// VideoMamba3D_35991825940955
// MI455X (gfx1250) — compile-verified, matrix operand order checked by reading, not run on hardware
//
#include <hip/hip_runtime.h>
#include <hip/hip_bf16.h>

// ---------------- problem constants ----------------
#define NBL  6
#define NN   2
#define LL   16
#define HH   14
#define WW   14
#define DD   192
#define DIN  384           // DI
#define DIX2 768           // 2*DI
#define DST  16            // DS (state dim)
#define DRR  12            // DR
#define KC   4             // conv kernel
#define SEQ  (LL*HH*WW)    // 3136
#define ROWS (NN*SEQ)      // 6272
#define XCOL (DRR + 2*DST) // 44

typedef __attribute__((ext_vector_type(16))) _Float16 v16h;
typedef __attribute__((ext_vector_type(8)))  float    v8f;

// ---------------- pack f32 weight B[K,N] into WMMA fragment order (f16) ----------------
// dst layout: [kt][nt][lane 0..31][e 0..15], fragment = contiguous 32 bytes per lane.
// CDNA5 16-bit B 32x16 layout: lane = hi*16 + n, elems 0..7 = K (kb..kb+7), 8..15 = K (16+kb..16+kb+7),
// kb = hi?8:0 with hi = lane>>4.
__global__ void pack_b_kernel(const float* __restrict__ src, _Float16* __restrict__ dst,
                              int Ncols, int Kdim) {
  int idx = blockIdx.x * blockDim.x + threadIdx.x;
  if (idx >= Kdim * Ncols) return;
  int e    = idx & 15;
  int lane = (idx >> 4) & 31;
  int tile = idx >> 9;
  int tilesN = Ncols >> 4;
  int nt = tile % tilesN;
  int kt = tile / tilesN;
  int r  = lane & 15;
  int hi = lane >> 4;
  int kb = hi ? 8 : 0;
  int ke = (e < 8) ? (kb + e) : (16 + kb + (e - 8));
  int k  = kt * 32 + ke;
  int n  = nt * 16 + r;
  dst[idx] = (_Float16)src[(size_t)k * Ncols + n];
}

// ---------------- LayerNorm -> f16 (one wave per row of 192) ----------------
__global__ void ln_kernel(const float* __restrict__ x, const float* __restrict__ w,
                          const float* __restrict__ b, _Float16* __restrict__ out) {
  int row = blockIdx.x * (blockDim.x >> 5) + (threadIdx.x >> 5);
  if (row >= ROWS) return;
  int lane = threadIdx.x & 31;
  const float* xr = x + (size_t)row * DD;
  float vals[6]; float s = 0.f, s2 = 0.f;
#pragma unroll
  for (int i = 0; i < 6; i++) { float v = xr[lane + 32*i]; vals[i] = v; s += v; s2 += v*v; }
#pragma unroll
  for (int off = 16; off >= 1; off >>= 1) { s += __shfl_xor(s, off, 32); s2 += __shfl_xor(s2, off, 32); }
  float mu = s * (1.0f / DD);
  float var = s2 * (1.0f / DD) - mu * mu;
  float rs = rsqrtf(var + 1e-5f);
#pragma unroll
  for (int i = 0; i < 6; i++) {
    int d = lane + 32*i;
    out[(size_t)row * DD + d] = (_Float16)((vals[i] - mu) * rs * w[d] + b[d]);
  }
}

// ---------------- WMMA GEMM: D = A(f16,[M,K]) * Bpacked (+ Cres if ADD) ----------------
// one wave32 per 16x16 output tile; K fully unrolled (KDIM/32 wmma issues per wave)
template <int KDIM, bool ADD>
__global__ void wmma_gemm_kernel(const _Float16* __restrict__ A, const _Float16* __restrict__ Bp,
                                 const float* __restrict__ Cres, float* __restrict__ Dst,
                                 int M, int Ncols) {
  int wave = (blockIdx.x * blockDim.x + threadIdx.x) >> 5;
  int tilesN = Ncols >> 4;
  int totTiles = (M >> 4) * tilesN;
  if (wave >= totTiles) return;
  int tm  = (wave / tilesN) << 4;
  int tnt = wave % tilesN;
  int lane = threadIdx.x & 31;
  int r  = lane & 15;
  int hi = lane >> 4;
  int kb = hi ? 8 : 0;

  v8f acc = {};
  const _Float16* Arow  = A + (size_t)(tm + r) * KDIM;
  const _Float16* Bfrag = Bp + ((size_t)tnt * 32 + lane) * 16;
  const size_t bstride  = (size_t)tilesN * 512;   // 32 lanes * 16 halves per k-tile row
#pragma unroll
  for (int kt = 0; kt < KDIM / 32; kt++) {
    const int k0 = kt * 32;
    __builtin_prefetch(Arow + k0 + 64, 0, 0);
    v16h af;
#pragma unroll
    for (int e = 0; e < 8; e++) af[e]     = Arow[k0 + kb + e];
#pragma unroll
    for (int e = 0; e < 8; e++) af[8 + e] = Arow[k0 + 16 + kb + e];
    v16h bf = *(const v16h*)(Bfrag + (size_t)kt * bstride);
    acc = __builtin_amdgcn_wmma_f32_16x16x32_f16(false, af, false, bf, (short)0, acc, false, false);
  }
  int n  = (tnt << 4) + r;
  int m0 = tm + (hi ? 8 : 0);
#pragma unroll
  for (int e = 0; e < 8; e++) {
    size_t o = (size_t)(m0 + e) * Ncols + n;
    if (ADD) Dst[o] = Cres[o] + acc[e];
    else     Dst[o] = acc[e];
  }
}

// ---------------- depthwise causal conv (K=4) + SiLU; rev flips direction ----------------
__global__ void conv_silu_kernel(const float* __restrict__ xz, const float* __restrict__ cw,
                                 const float* __restrict__ cb, float* __restrict__ xc, int rev) {
  int idx = blockIdx.x * blockDim.x + threadIdx.x;
  if (idx >= ROWS * DIN) return;
  int d = idx % DIN;
  int row = idx / DIN;
  int s = row % SEQ;
  int b = row / SEQ;
  float acc = cb[d];
  int dir = rev ? -1 : 1;
#pragma unroll
  for (int k = 0; k < KC; k++) {
    int o = s + dir * (k - (KC - 1));
    if (o >= 0 && o < SEQ)
      acc += xz[(size_t)(b * SEQ + o) * DIX2 + d] * cw[d * KC + k];
  }
  xc[(size_t)row * DIN + d] = acc * (1.0f / (1.0f + __expf(-acc)));
}

// ---------------- xdbl = xc @ Wx  (K=384, N=44; small N -> plain kernel) ----------------
__global__ void gemm_wx_kernel(const float* __restrict__ xc, const float* __restrict__ Wx,
                               float* __restrict__ xdbl) {
  int idx = blockIdx.x * blockDim.x + threadIdx.x;
  if (idx >= ROWS * XCOL) return;
  int j = idx % XCOL;
  int row = idx / XCOL;
  const float* xr = xc + (size_t)row * DIN;
  float acc = 0.f;
  for (int i = 0; i < DIN; i++) acc += xr[i] * Wx[i * XCOL + j];
  xdbl[(size_t)row * XCOL + j] = acc;
}

// ---------------- dt = softplus(xdbl[:,:12] @ Wdt + bdt) ----------------
__global__ void dt_kernel(const float* __restrict__ xdbl, const float* __restrict__ Wdt,
                          const float* __restrict__ bdt, float* __restrict__ dtb) {
  int idx = blockIdx.x * blockDim.x + threadIdx.x;
  if (idx >= ROWS * DIN) return;
  int d = idx % DIN;
  int row = idx / DIN;
  const float* xr = xdbl + (size_t)row * XCOL;
  float acc = bdt[d];
#pragma unroll
  for (int r = 0; r < DRR; r++) acc += xr[r] * Wdt[r * DIN + d];
  float sp = (acc > 20.f) ? acc : log1pf(__expf(acc));
  dtb[(size_t)row * DIN + d] = sp;
}

// ---------------- selective-scan: 2 chains per wave, 16 state lanes per chain ----------------
__global__ void scan_kernel(const float* __restrict__ dtb, const float* __restrict__ xdbl,
                            const float* __restrict__ xc, const float* __restrict__ xz,
                            const float* __restrict__ A_log, const float* __restrict__ Dp,
                            _Float16* __restrict__ yh, int rev) {
  int wave = (blockIdx.x * blockDim.x + threadIdx.x) >> 5;
  int lane = threadIdx.x & 31;
  int chain = wave * 2 + (lane >> 4);
  if (chain >= NN * DIN) return;
  int n = lane & 15;
  int b = chain / DIN;
  int d = chain % DIN;
  float Ac = -__expf(A_log[d * DST + n]);
  float dp = Dp[d];
  float st = 0.f;
  for (int t = 0; t < SEQ; t++) {
    int s = rev ? (SEQ - 1 - t) : t;
    size_t row = (size_t)b * SEQ + s;
    float dtv = dtb[row * DIN + d];
    float xcv = xc[row * DIN + d];
    float Bn  = xdbl[row * XCOL + DRR + n];
    float Cn  = xdbl[row * XCOL + DRR + DST + n];
    st = __expf(dtv * Ac) * st + dtv * Bn * xcv;
    float contrib = st * Cn;
#pragma unroll
    for (int off = 8; off >= 1; off >>= 1) contrib += __shfl_xor(contrib, off, 32);
    if (n == 0) {
      float y = contrib + xcv * dp;
      float z = xz[row * DIX2 + DIN + d];
      y *= z * (1.0f / (1.0f + __expf(-z)));
      yh[row * DIN + d] = (_Float16)y;
    }
  }
}

// ---------------- sequence-order permutations ----------------
// order A: sA = ((h*W + w)*L + l)   (layers 0,1)
// order B: sB = ((l*W + w)*H + h)   (layers 2,3)
// order C: sC = ((l*H + h)*W + w)   (layers 4,5; == canonical (N,L,H,W,D) flatten)
__global__ void gather_in_kernel(const float* __restrict__ x, float* __restrict__ dst) {
  int idx = blockIdx.x * blockDim.x + threadIdx.x;
  if (idx >= ROWS * DD) return;
  int d = idx % DD;
  int row = idx / DD;
  int sA = row % SEQ;
  int b = row / SEQ;
  int l = sA % LL;
  int w = (sA / LL) % WW;
  int h = sA / (LL * WW);
  size_t src = ((((size_t)b * LL + l) * HH + h) * WW + w) * DD + d;
  dst[idx] = x[src];
}

__global__ void perm_AB_kernel(const float* __restrict__ src, float* __restrict__ dst) {
  int idx = blockIdx.x * blockDim.x + threadIdx.x;
  if (idx >= ROWS * DD) return;
  int d = idx % DD;
  int row = idx / DD;
  int sB = row % SEQ;
  int b = row / SEQ;
  int h = sB % HH;
  int w = (sB / HH) % WW;
  int l = sB / (HH * WW);
  int sA = (h * WW + w) * LL + l;
  dst[idx] = src[(size_t)(b * SEQ + sA) * DD + d];
}

__global__ void perm_BC_kernel(const float* __restrict__ src, float* __restrict__ dst) {
  int idx = blockIdx.x * blockDim.x + threadIdx.x;
  if (idx >= ROWS * DD) return;
  int d = idx % DD;
  int row = idx / DD;
  int sC = row % SEQ;
  int b = row / SEQ;
  int w = sC % WW;
  int h = (sC / WW) % HH;
  int l = sC / (WW * HH);
  int sB = (l * WW + w) * HH + h;
  dst[idx] = src[(size_t)(b * SEQ + sB) * DD + d];
}

// ---------------- host launch ----------------
extern "C" void kernel_launch(void* const* d_in, const int* in_sizes, int n_in,
                              void* d_out, int out_size, void* d_ws, size_t ws_size,
                              hipStream_t stream) {
  const float* x     = (const float*)d_in[0];
  const float* ln_w  = (const float*)d_in[1];
  const float* ln_b  = (const float*)d_in[2];
  const float* Win   = (const float*)d_in[3];
  const float* cw    = (const float*)d_in[4];
  const float* cb    = (const float*)d_in[5];
  const float* Wx    = (const float*)d_in[6];
  const float* Wdt   = (const float*)d_in[7];
  const float* bdt   = (const float*)d_in[8];
  const float* A_log = (const float*)d_in[9];
  const float* Dp    = (const float*)d_in[10];
  const float* Wout  = (const float*)d_in[11];
  float* out = (float*)d_out;

  char* p = (char*)d_ws;
  auto alloc = [&](size_t bytes) -> char* {
    char* q = p;
    p += (bytes + 255) & ~(size_t)255;
    return q;
  };
  float*     bufA   = (float*)    alloc((size_t)ROWS * DD * 4);
  float*     bufB   = (float*)    alloc((size_t)ROWS * DD * 4);
  _Float16*  h_h    = (_Float16*) alloc((size_t)ROWS * DD * 2);
  float*     xzb    = (float*)    alloc((size_t)ROWS * DIX2 * 4);
  float*     xcb    = (float*)    alloc((size_t)ROWS * DIN * 4);
  float*     xdblb  = (float*)    alloc((size_t)ROWS * XCOL * 4);
  float*     dtb    = (float*)    alloc((size_t)ROWS * DIN * 4);
  _Float16*  y_h    = (_Float16*) alloc((size_t)ROWS * DIN * 2);
  _Float16*  Win_p  = (_Float16*) alloc((size_t)DD * DIX2 * 2);   // packed fragments
  _Float16*  Wout_p = (_Float16*) alloc((size_t)DIN * DD * 2);    // packed fragments

  const int TB = 256;
  auto blks = [](long n, int tb) { return (int)((n + tb - 1) / tb); };

  // gather canonical (N,L,H,W,D) input into scan-order A
  gather_in_kernel<<<blks((long)ROWS * DD, TB), TB, 0, stream>>>(x, bufA);

  float* cur = bufA;
  for (int layer = 0; layer < NBL; layer++) {
    int rev = layer & 1;

    // pack weights into WMMA fragment order (f16)
    pack_b_kernel<<<blks(DD * DIX2, TB), TB, 0, stream>>>(Win + (size_t)layer * DD * DIX2, Win_p, DIX2, DD);
    pack_b_kernel<<<blks(DIN * DD, TB), TB, 0, stream>>>(Wout + (size_t)layer * DIN * DD, Wout_p, DD, DIN);

    // 1) LayerNorm -> f16 activations
    ln_kernel<<<blks(ROWS, TB / 32), TB, 0, stream>>>(cur, ln_w + layer * DD, ln_b + layer * DD, h_h);

    // 2) xz = h @ Win  (WMMA, M=6272 K=192 N=768; 6 unrolled wmma per tile)
    {
      int tiles = (ROWS / 16) * (DIX2 / 16);
      wmma_gemm_kernel<DD, false><<<blks((long)tiles * 32, TB), TB, 0, stream>>>(
          h_h, Win_p, nullptr, xzb, ROWS, DIX2);
    }

    // 3) depthwise causal conv + SiLU (direction-aware)
    conv_silu_kernel<<<blks((long)ROWS * DIN, TB), TB, 0, stream>>>(
        xzb, cw + (size_t)layer * DIN * KC, cb + (size_t)layer * DIN, xcb, rev);

    // 4) xdbl = xc @ Wx  (N=44)
    gemm_wx_kernel<<<blks((long)ROWS * XCOL, TB), TB, 0, stream>>>(xcb, Wx + (size_t)layer * DIN * XCOL, xdblb);

    // 5) dt = softplus(xdbl[:, :12] @ Wdt + bdt)
    dt_kernel<<<blks((long)ROWS * DIN, TB), TB, 0, stream>>>(
        xdblb, Wdt + (size_t)layer * DRR * DIN, bdt + (size_t)layer * DIN, dtb);

    // 6) selective scan + skip term + SiLU(z) gate -> f16
    {
      int chains = NN * DIN;              // 768 chains, 2 per wave
      int threads = (chains / 2) * 32;    // 12288
      scan_kernel<<<blks(threads, TB), TB, 0, stream>>>(
          dtb, xdblb, xcb, xzb, A_log + (size_t)layer * DIN * DST, Dp + (size_t)layer * DIN, y_h, rev);
    }

    // 7) residual: dst = cur + y_gated @ Wout  (WMMA, K=384 N=192; 12 unrolled wmma per tile)
    {
      float* dst = (layer == NBL - 1) ? out : cur;
      int tiles = (ROWS / 16) * (DD / 16);
      wmma_gemm_kernel<DIN, true><<<blks((long)tiles * 32, TB), TB, 0, stream>>>(
          y_h, Wout_p, cur, dst, ROWS, DD);
    }

    // reorder sequence between layer pairs
    if (layer == 1) {
      perm_AB_kernel<<<blks((long)ROWS * DD, TB), TB, 0, stream>>>(bufA, bufB);
      cur = bufB;
    } else if (layer == 3) {
      perm_BC_kernel<<<blks((long)ROWS * DD, TB), TB, 0, stream>>>(bufB, bufA);
      cur = bufA;
    }
  }
}
